// KmeansVectorQuantizer_27779848470626
// MI455X (gfx1250) — matrix operand order repaired and checked by reading, not compiled, not run on hardware
//
#include <hip/hip_runtime.h>
#include <hip/hip_bf16.h>
#include <math.h>

// ---------------- problem constants ----------------
#define B_  16
#define T_  2048
#define C_  512
#define G_  2
#define DV_ 256
#define V_  320

// ---------------- WMMA types (gfx1250, wave32) ----------------
typedef __attribute__((ext_vector_type(16))) __bf16 v16bf;
typedef __attribute__((ext_vector_type(8)))  float  v8f;

// ---------------- workspace layout (bytes) ----------------
static constexpr size_t ZE_OFF    = 0;                                      // f32  B*G*T*DV  = 64 MB
static constexpr size_t WT_OFF    = (size_t)B_*G_*T_*DV_*4;                 // bf16 G*DV*DV   (wt[g][o][k])
static constexpr size_t EMBT_OFF  = WT_OFF   + (size_t)G_*DV_*DV_*2;        // bf16 G*V*DV    (emb2[g][v][d])
static constexpr size_t EMBN_OFF  = EMBT_OFF + (size_t)G_*V_*DV_*2;         // f32  G*V
static constexpr size_t STATS_OFF = EMBN_OFF + (size_t)G_*V_*4;             // f32  B*G*2 (sum,sumsq)
static constexpr size_t MUSIG_OFF = STATS_OFF + (size_t)B_*G_*2*4;          // f32  B*G*2 (mu,rsig)
static constexpr size_t HIST_OFF  = MUSIG_OFF + (size_t)B_*G_*2*4;          // f32  G*V
static constexpr size_t LOSSA_OFF = HIST_OFF + (size_t)G_*V_*4;             // f32  1

// ---------------- output layout (floats), tuple order (out, loss, ppl, idx) ----------------
static constexpr size_t OUT_N    = (size_t)B_*T_*C_;   // 16777216
static constexpr size_t LOSS_POS = OUT_N;
static constexpr size_t PPL_POS  = OUT_N + 1;
static constexpr size_t IDX_POS  = OUT_N + 2;          // B*T*G entries (stored as float)

// =======================================================================
// Prep: conv_w -> bf16 ([g][o][k], K contiguous = contiguous B fragments);
// emb -> emb2[g][v][d] bf16 (d contiguous); ||emb||^2.
// =======================================================================
__global__ void vq_prep(const float* __restrict__ conv_w,
                        const float* __restrict__ emb,
                        __bf16* __restrict__ wt,
                        __bf16* __restrict__ emb2,
                        float*  __restrict__ embn) {
  int tid = blockIdx.x * blockDim.x + threadIdx.x;
  int nth = gridDim.x * blockDim.x;
  for (int i = tid; i < G_*DV_*DV_; i += nth) {
    wt[i] = (__bf16)conv_w[i];
  }
  for (int i = tid; i < G_*V_*DV_; i += nth) {
    int g = i / (V_*DV_); int r = i % (V_*DV_);
    int v = r / DV_;      int d = r % DV_;
    emb2[i] = (__bf16)emb[((size_t)v*G_ + g)*DV_ + d];
  }
  for (int i = tid; i < G_*V_; i += nth) {
    int g = i / V_; int v = i % V_;
    const float* e = emb + ((size_t)v*G_ + g)*DV_;
    float s = 0.f;
    for (int d = 0; d < DV_; ++d) { float t = e[d]; s += t*t; }
    embn[i] = s;
  }
}

// =======================================================================
// Conv GEMM: ze[(b,g,t),o] = sum_k x[b,t,g*256+k] * W[g][o][k]
// One wave owns a 16(t) x 64(o) slab: A fragments built ONCE in VGPRs,
// 4 concurrent o-tile accumulators (no WMMA->WMMA hazards) with
// software-pipelined (double-buffered) B-fragment loads.
// =======================================================================
__global__ void __launch_bounds__(256) vq_conv(const float* __restrict__ x,
                                               const __bf16* __restrict__ wt,
                                               float* __restrict__ ze,
                                               float* __restrict__ stats) {
  const int wave = (blockIdx.x * blockDim.x + threadIdx.x) >> 5;  // 16384 waves
  const int lane = threadIdx.x & 31;
  const int half = lane >> 4;
  const int nrow = lane & 15;
  int op = wave & 3;  int tmp = wave >> 2;     // o-slab (4 tiles of 16)
  int tt = tmp & 127; tmp >>= 7;
  int g  = tmp & 1;   int b = tmp >> 1;
  const int t0 = tt * 16;

  // Build A fragments once (row m = nrow): one contiguous 16-float run/chunk.
  const float* arow = x + ((size_t)b*T_ + t0 + nrow)*C_ + g*DV_;
  v16bf afrag[8];
#pragma unroll
  for (int kc = 0; kc < 8; ++kc) {
    const int k0 = kc * 32;
#pragma unroll
    for (int i = 0; i < 8; ++i) {
      afrag[kc][i]     = (__bf16)arow[k0 + 16*half + i];
      afrag[kc][8 + i] = (__bf16)arow[k0 + 16*half + 8 + i];
    }
  }

  const __bf16* brow[4];
#pragma unroll
  for (int j = 0; j < 4; ++j)
    brow[j] = wt + ((size_t)g*DV_ + (op*4 + j)*16 + nrow)*DV_;

  v8f acc[4];
#pragma unroll
  for (int j = 0; j < 4; ++j) acc[j] = (v8f){0.f,0.f,0.f,0.f,0.f,0.f,0.f,0.f};

  // software-pipelined B fragments: issue chunk kc+1 loads before chunk kc WMMAs
  v16bf cb[4];
#pragma unroll
  for (int j = 0; j < 4; ++j) cb[j] = *(const v16bf*)(brow[j] + 16*half);
#pragma unroll
  for (int kc = 0; kc < 8; ++kc) {
    v16bf nb[4];
    if (kc < 7) {
      const int kn = (kc + 1) * 32;
#pragma unroll
      for (int j = 0; j < 4; ++j) nb[j] = *(const v16bf*)(brow[j] + kn + 16*half);
    }
#pragma unroll
    for (int j = 0; j < 4; ++j)
      acc[j] = __builtin_amdgcn_wmma_f32_16x16x32_bf16(false, afrag[kc], false, cb[j],
                                                       (short)0, acc[j], false, false);
    if (kc < 7) {
#pragma unroll
      for (int j = 0; j < 4; ++j) cb[j] = nb[j];
    }
  }

  // D layout: acc[r] = D[r + 8*half][lane%16]
  float s1 = 0.f, s2 = 0.f;
  size_t zbase = (((size_t)(b*G_ + g))*T_ + t0)*DV_ + op*64 + nrow;
#pragma unroll
  for (int j = 0; j < 4; ++j) {
#pragma unroll
    for (int r = 0; r < 8; ++r) {
      float v = acc[j][r];
      ze[zbase + (size_t)(r + 8*half)*DV_ + j*16] = v;
      s1 += v; s2 += v*v;
    }
  }
  // wave-level reduction, then 2 atomics per wave
#pragma unroll
  for (int off = 16; off >= 1; off >>= 1) {
    s1 += __shfl_xor(s1, off, 32);
    s2 += __shfl_xor(s2, off, 32);
  }
  if (lane == 0) {
    atomicAdd(&stats[(b*G_ + g)*2 + 0], s1);
    atomicAdd(&stats[(b*G_ + g)*2 + 1], s2);
  }
}

// =======================================================================
// Finalize GroupNorm statistics: mu, rsqrt(var+eps) per (b,g)
// =======================================================================
__global__ void vq_stats(const float* __restrict__ stats, float* __restrict__ musig) {
  int i = threadIdx.x;
  if (i < B_*G_) {
    const float n = (float)DV_ * (float)T_;
    float mu  = stats[i*2] / n;
    float var = stats[i*2+1] / n - mu*mu;
    musig[i*2]   = mu;
    musig[i*2+1] = rsqrtf(var + 1e-5f);
  }
}

// =======================================================================
// Distance GEMM + argmin + gather + loss + histogram + idx
// One wave per (b,g, t-tile of 16). A = normalized ze rows (bf16, built once,
// kept in VGPRs). 10 iterations of PAIRED codebook tiles, 2 independent
// accumulators, double-buffered B loads. Argmin w/ first-index tie-break.
// =======================================================================
__global__ void __launch_bounds__(256) vq_dist(const float* __restrict__ ze,
                                               const __bf16* __restrict__ emb2,
                                               const float* __restrict__ embn,
                                               const float* __restrict__ musig,
                                               const float* __restrict__ gn_w,
                                               const float* __restrict__ gn_b,
                                               const float* __restrict__ emb,
                                               float* __restrict__ out,
                                               float* __restrict__ hist,
                                               float* __restrict__ lossacc) {
  __shared__ float lds_rn[8][16];
  __shared__ int   lds_bv[8][16];
  const int w    = threadIdx.x >> 5;
  const int lane = threadIdx.x & 31;
  const int half = lane >> 4;
  const int nrow = lane & 15;
  int wave = (blockIdx.x * blockDim.x + threadIdx.x) >> 5;
  int tt = wave & 127; int tmp = wave >> 7;
  int g  = tmp & 1;    int b = tmp >> 1;
  const int t0 = tt * 16;
  const int bg = b*G_ + g;

  const float mu = musig[bg*2], rs = musig[bg*2+1];
  const float* gwp = gn_w + g*DV_;
  const float* gbp = gn_b + g*DV_;

  // Build normalized A fragments (row m = nrow), accumulate f32 row norm.
  size_t zrow = (((size_t)bg)*T_ + t0 + nrow)*DV_;
  v16bf afrag[8];
  float rnp = 0.f;
#pragma unroll
  for (int kc = 0; kc < 8; ++kc) {
    const int k0 = kc * 32;
#pragma unroll
    for (int i = 0; i < 8; ++i) {
      int ka = k0 + 16*half + i;
      int kb = k0 + 16*half + 8 + i;
      float z0 = (ze[zrow + ka] - mu)*rs*gwp[ka] + gbp[ka];
      float z1 = (ze[zrow + kb] - mu)*rs*gwp[kb] + gbp[kb];
      afrag[kc][i]     = (__bf16)z0;
      afrag[kc][8 + i] = (__bf16)z1;
      rnp += z0*z0 + z1*z1;
    }
  }
  float rn = rnp + __shfl_xor(rnp, 16, 32);   // combine the two K-halves of row nrow
  if (half == 0) lds_rn[w][nrow] = rn;
  __builtin_amdgcn_wave_barrier();

  float bestd[8]; int bestv[8];
#pragma unroll
  for (int r = 0; r < 8; ++r) { bestd[r] = 3.4e38f; bestv[r] = 0; }

  const __bf16* bbase = emb2 + (size_t)g*V_*DV_;
  for (int vt = 0; vt < V_/32; ++vt) {        // paired tiles: v0 and v0+16
    const int v0 = vt * 32;
    const __bf16* brow0 = bbase + ((size_t)(v0 + nrow))*DV_;
    const __bf16* brow1 = brow0 + (size_t)16*DV_;
    if (vt + 1 < V_/32)
      __builtin_prefetch(brow0 + (size_t)32*DV_, 0, 0);   // next pair -> global_prefetch_b8
    v8f acc0 = {0.f,0.f,0.f,0.f,0.f,0.f,0.f,0.f};
    v8f acc1 = {0.f,0.f,0.f,0.f,0.f,0.f,0.f,0.f};
    // double-buffered B fragments
    v16bf b0 = *(const v16bf*)(brow0 + 16*half);
    v16bf b1 = *(const v16bf*)(brow1 + 16*half);
#pragma unroll
    for (int kc = 0; kc < 8; ++kc) {
      v16bf n0, n1;
      if (kc < 7) {
        const int kn = (kc + 1) * 32;
        n0 = *(const v16bf*)(brow0 + kn + 16*half);
        n1 = *(const v16bf*)(brow1 + kn + 16*half);
      }
      acc0 = __builtin_amdgcn_wmma_f32_16x16x32_bf16(false, afrag[kc], false, b0,
                                                     (short)0, acc0, false, false);
      acc1 = __builtin_amdgcn_wmma_f32_16x16x32_bf16(false, afrag[kc], false, b1,
                                                     (short)0, acc1, false, false);
      if (kc < 7) { b0 = n0; b1 = n1; }
    }
    float en0 = embn[g*V_ + v0 + nrow];
    float en1 = embn[g*V_ + v0 + 16 + nrow];
#pragma unroll
    for (int r = 0; r < 8; ++r) {
      float rnr = lds_rn[w][r + 8*half];
      float d20 = rnr - 2.f*acc0[r] + en0;
      float d21 = rnr - 2.f*acc1[r] + en1;
      if (d20 < bestd[r]) { bestd[r] = d20; bestv[r] = v0 + nrow; }       // lower v first
      if (d21 < bestd[r]) { bestd[r] = d21; bestv[r] = v0 + 16 + nrow; }
    }
  }

  // Cross-lane argmin within each 16-lane half (rows r+8*half), first-index tie-break.
#pragma unroll
  for (int r = 0; r < 8; ++r) {
    float d = bestd[r]; int v = bestv[r];
#pragma unroll
    for (int off = 8; off >= 1; off >>= 1) {
      float od = __shfl_xor(d, off, 32);
      int   ov = __shfl_xor(v, off, 32);
      if (od < d || (od == d && ov < v)) { d = od; v = ov; }
    }
    if (nrow == 0) {
      int row = r + 8*half;
      lds_bv[w][row] = v;
      out[IDX_POS + ((size_t)(b*T_ + t0 + row))*G_ + g] = (float)v;  // idx as float
      atomicAdd(&hist[g*V_ + v], 1.0f);
    }
  }
  __builtin_amdgcn_wave_barrier();

  // Cooperative float4 gather of emb[idx] -> out + loss partial
  // (ze rows are L2-resident on the 192MB L2; re-read is cheap).
  float lacc = 0.f;
  const float4* gw4 = (const float4*)gwp;
  const float4* gb4 = (const float4*)gbp;
  for (int row = 0; row < 16; ++row) {
    int bv = lds_bv[w][row];
    const float4* zr4 = (const float4*)(ze + (((size_t)bg)*T_ + t0 + row)*DV_);
    const float4* er4 = (const float4*)(emb + ((size_t)bv*G_ + g)*DV_);
    float4*       ot4 = (float4*)(out + ((size_t)b*T_ + t0 + row)*C_ + g*DV_);
#pragma unroll
    for (int i = 0; i < 2; ++i) {
      int q = lane*2 + i;                       // float4 index 0..63
      float4 z = zr4[q], e = er4[q], wv = gw4[q], bvv = gb4[q];
      float d0 = e.x - ((z.x - mu)*rs*wv.x + bvv.x);
      float d1 = e.y - ((z.y - mu)*rs*wv.y + bvv.y);
      float d2 = e.z - ((z.z - mu)*rs*wv.z + bvv.z);
      float d3 = e.w - ((z.w - mu)*rs*wv.w + bvv.w);
      lacc += d0*d0 + d1*d1 + d2*d2 + d3*d3;
      ot4[q] = e;                               // forward value of ST output == zq
    }
  }
  atomicAdd(lossacc, lacc);
}

// =======================================================================
// Scalars: loss = 1.25 * mean((zq-zn)^2); perplexity from (G,V) histogram
// =======================================================================
__global__ void vq_fin(const float* __restrict__ hist,
                       const float* __restrict__ lossacc,
                       float* __restrict__ out) {
  if (threadIdx.x == 0 && blockIdx.x == 0) {
    out[LOSS_POS] = lossacc[0] * (1.25f / ((float)B_*(float)C_*(float)T_));
    float ppl = 0.f;
    for (int g = 0; g < G_; ++g) {
      float ent = 0.f;
      for (int v = 0; v < V_; ++v) {
        float p = hist[g*V_ + v] * (1.0f / ((float)B_*(float)T_));
        ent += p * logf(p + 1e-7f);
      }
      ppl += expf(-ent);
    }
    out[PPL_POS] = ppl;
  }
}

// =======================================================================
extern "C" void kernel_launch(void* const* d_in, const int* in_sizes, int n_in,
                              void* d_out, int out_size, void* d_ws, size_t ws_size,
                              hipStream_t stream) {
  const float* x      = (const float*)d_in[0];
  const float* conv_w = (const float*)d_in[1];
  const float* gn_w   = (const float*)d_in[2];
  const float* gn_b   = (const float*)d_in[3];
  const float* emb    = (const float*)d_in[4];
  float* out = (float*)d_out;
  char*  ws  = (char*)d_ws;

  float*  ze      = (float*)(ws + ZE_OFF);
  __bf16* wt      = (__bf16*)(ws + WT_OFF);
  __bf16* emb2    = (__bf16*)(ws + EMBT_OFF);
  float*  embn    = (float*)(ws + EMBN_OFF);
  float*  stats   = (float*)(ws + STATS_OFF);
  float*  musig   = (float*)(ws + MUSIG_OFF);
  float*  hist    = (float*)(ws + HIST_OFF);
  float*  lossacc = (float*)(ws + LOSSA_OFF);

  // zero the accumulators (stats..lossacc are contiguous) — capturable stream op
  hipMemsetAsync(ws + STATS_OFF, 0, (LOSSA_OFF + sizeof(float)) - STATS_OFF, stream);

  vq_prep <<<512, 256, 0, stream>>>(conv_w, emb, wt, emb2, embn);
  vq_conv <<<2048, 256, 0, stream>>>(x, wt, ze, stats);        // 16384 waves, 16x64 slabs
  vq_stats<<<1, 32, 0, stream>>>(stats, musig);
  vq_dist <<<512, 256, 0, stream>>>(ze, emb2, embn, musig, gn_w, gn_b, emb,
                                    out, hist, lossacc);       // 4096 wave-tiles
  vq_fin  <<<1, 1, 0, stream>>>(hist, lossacc, out);
}
